// RealQuantLinear_65206193488391
// MI455X (gfx1250) — matrix operand order repaired and checked by reading, not compiled, not run on hardware
//
#include <hip/hip_runtime.h>
#include <hip/hip_bf16.h>

typedef __attribute__((ext_vector_type(16))) _Float16 v16h;
typedef __attribute__((ext_vector_type(8)))  _Float16 v8h;
typedef __attribute__((ext_vector_type(8)))  float    v8f;
typedef __attribute__((ext_vector_type(4)))  unsigned v4u;

// Problem dims (fixed by the reference)
constexpr int M_TOTAL = 4 * 2048;   // B*S = 8192
constexpr int N_TOTAL = 11008;      // OUT
constexpr int K_TOTAL = 4096;       // IN

// Tiling: block 128(M) x 256(N), 8 waves as 2(M) x 4(N), wave tile 64x64 = 4x4 fragments
constexpr int TILE_M = 128;
constexpr int TILE_N = 256;
constexpr int TILE_K = 32;          // one WMMA K-depth
constexpr int LDSS   = TILE_K + 8;  // padded row stride in halves (80B) -> conflict-free ds_load_b128

// Magic-bias dequant: byte w holds two int4 nibbles (lo = even K, hi = odd K).
// Place (nibble ^ 8) = q+8 in mantissa LSBs of f16 1024.0 -> value 1032+q; subtract 1032.
__device__ __forceinline__ unsigned dq2_bits(int w) {
    unsigned t = ((unsigned)w & 0xFu) | (((unsigned)w << 12) & 0x000F0000u);
    t ^= 0x00080008u;      // nibble^8 = q+8 in [0,15]
    t |= 0x64006400u;      // f16 1024.0 in both halves
    return t;              // each half = 1032 + q (exact in f16)
}

__global__ __launch_bounds__(256)
void q4_gemm_wmma(const float* __restrict__ x,
                  const int*   __restrict__ wq,     // [N, K/2] one packed byte per int
                  const float* __restrict__ scale,  // [N]
                  const float* __restrict__ bias,   // [N]
                  float*       __restrict__ out)    // [M, N]
{
    // double-buffered tiles: 2 * (10.0 + 20.0) KB = 60 KB
    __shared__ alignas(16) _Float16 lds_a[2][TILE_M * LDSS];
    __shared__ alignas(16) _Float16 lds_b[2][TILE_N * LDSS];

    const int tid = threadIdx.x;
    const int n0  = blockIdx.x * TILE_N;
    const int m0  = blockIdx.y * TILE_M;

    const int wave  = tid >> 5;
    const int lane  = tid & 31;
    const int lhalf = lane >> 4;
    const int lmod  = lane & 15;
    const int wm    = wave & 1;     // wave M position: wm*64
    const int wn    = wave >> 1;    // wave N position: wn*64

    // per-thread staging slice (fixed across K-steps)
    const int arow[2]  = { (tid + 0)   >> 2, (tid + 256) >> 2 };          // 0..127
    const int acol8[2] = { ((tid + 0)  & 3) * 8, ((tid + 256) & 3) * 8 }; // 8-float segment
    int brow[4], bcg[4];
    #pragma unroll
    for (int i = 0; i < 4; ++i) { brow[i] = (tid + i * 256) >> 2; bcg[i] = (tid + i * 256) & 3; }

    // ---- staging helpers ----
    auto load_tile = [&](int k0, float4 va[2][2], int4 vb[4]) {
        #pragma unroll
        for (int i = 0; i < 2; ++i) {
            const float* src = x + (size_t)(m0 + arow[i]) * K_TOTAL + k0 + acol8[i];
            va[i][0] = *(const float4*)(src);
            va[i][1] = *(const float4*)(src + 4);
        }
        #pragma unroll
        for (int i = 0; i < 4; ++i)
            vb[i] = *(const int4*)(wq + (size_t)(n0 + brow[i]) * (K_TOTAL / 2)
                                      + (k0 >> 1) + bcg[i] * 4);
    };
    auto store_tile = [&](int buf, const float4 va[2][2], const int4 vb[4]) {
        #pragma unroll
        for (int i = 0; i < 2; ++i) {
            v8h hv;
            hv[0] = (_Float16)va[i][0].x; hv[1] = (_Float16)va[i][0].y;
            hv[2] = (_Float16)va[i][0].z; hv[3] = (_Float16)va[i][0].w;
            hv[4] = (_Float16)va[i][1].x; hv[5] = (_Float16)va[i][1].y;
            hv[6] = (_Float16)va[i][1].z; hv[7] = (_Float16)va[i][1].w;
            *(v8h*)(&lds_a[buf][arow[i] * LDSS + acol8[i]]) = hv;
        }
        #pragma unroll
        for (int i = 0; i < 4; ++i) {
            const v4u bits = { dq2_bits(vb[i].x), dq2_bits(vb[i].y),
                               dq2_bits(vb[i].z), dq2_bits(vb[i].w) };
            v8h hv = __builtin_bit_cast(v8h, bits) - (_Float16)1032.0f;  // 4x v_pk_add_f16
            *(v8h*)(&lds_b[buf][brow[i] * LDSS + bcg[i] * 8]) = hv;
        }
    };

    v8f acc[4][4];
    #pragma unroll
    for (int mi = 0; mi < 4; ++mi)
        #pragma unroll
        for (int ni = 0; ni < 4; ++ni)
            #pragma unroll
            for (int r = 0; r < 8; ++r)
                acc[mi][ni][r] = 0.0f;

    // ---- pipeline prologue: stage K-tile 0 into buffer 0 ----
    float4 va[2][2]; int4 vb[4];
    load_tile(0, va, vb);
    store_tile(0, va, vb);
    __syncthreads();

    // ---- main pipeline: 1 barrier per K-step; next-tile loads overlap WMMA ----
    for (int k0 = 0; k0 < K_TOTAL; k0 += TILE_K) {
        const int  cur  = (k0 / TILE_K) & 1;
        const bool more = (k0 + TILE_K) < K_TOTAL;

        if (more) load_tile(k0 + TILE_K, va, vb);   // latency hidden under WMMAs

        // B fragments (32x16 f16): lanes 0-15 hold K=0-7,16-23 (N=lane), lanes 16-31 K=8-15,24-31
        const _Float16* bbase0 = &lds_b[cur][0];
        v16h bfrag[4];
        #pragma unroll
        for (int ni = 0; ni < 4; ++ni) {
            const _Float16* base = bbase0 + (wn * 64 + ni * 16 + lmod) * LDSS;
            v8h lo = *(const v8h*)(base + lhalf * 8);
            v8h hi = *(const v8h*)(base + 16 + lhalf * 8);
            bfrag[ni] = __builtin_shufflevector(lo, hi, 0,1,2,3,4,5,6,7,8,9,10,11,12,13,14,15);
        }
        // A fragment per mi (16x32 f16), 4 WMMAs each against resident B frags
        const _Float16* abase0 = &lds_a[cur][0];
        #pragma unroll
        for (int mi = 0; mi < 4; ++mi) {
            const _Float16* base = abase0 + (wm * 64 + mi * 16 + lmod) * LDSS;
            v8h lo = *(const v8h*)(base + lhalf * 8);
            v8h hi = *(const v8h*)(base + 16 + lhalf * 8);
            v16h afrag = __builtin_shufflevector(lo, hi, 0,1,2,3,4,5,6,7,8,9,10,11,12,13,14,15);
            #pragma unroll
            for (int ni = 0; ni < 4; ++ni)
                acc[mi][ni] = __builtin_amdgcn_wmma_f32_16x16x32_f16(
                    /*neg_a=*/false, afrag,
                    /*neg_b=*/false, bfrag[ni],
                    /*c_mod=*/(short)0, acc[mi][ni],
                    /*reuse_a=*/false, /*reuse_b=*/false);
        }

        if (more) store_tile(cur ^ 1, va, vb);      // fill the other buffer
        __syncthreads();
    }

    // ---- epilogue: D layout M = lhalf*8 + r, N = lmod; per-channel scale/bias in f32 ----
    #pragma unroll
    for (int ni = 0; ni < 4; ++ni) {
        const int n = n0 + wn * 64 + ni * 16 + lmod;
        const float s  = scale[n];
        const float bb = bias[n];
        #pragma unroll
        for (int mi = 0; mi < 4; ++mi) {
            #pragma unroll
            for (int r = 0; r < 8; ++r) {
                const int m = m0 + wm * 64 + mi * 16 + lhalf * 8 + r;
                out[(size_t)m * N_TOTAL + n] = acc[mi][ni][r] * s + bb;
            }
        }
    }
}

extern "C" void kernel_launch(void* const* d_in, const int* in_sizes, int n_in,
                              void* d_out, int out_size, void* d_ws, size_t ws_size,
                              hipStream_t stream) {
    const float* x     = (const float*)d_in[0];
    const int*   wq    = (const int*)d_in[1];
    const float* scale = (const float*)d_in[2];
    const float* bias  = (const float*)d_in[3];
    float* out = (float*)d_out;

    dim3 grid(N_TOTAL / TILE_N, M_TOTAL / TILE_M);   // 43 x 64
    q4_gemm_wmma<<<grid, dim3(256), 0, stream>>>(x, wq, scale, bias, out);
}
